// GatedGraphConvolution_29867202576581
// MI455X (gfx1250) — matrix operand order repaired and checked by reading, not compiled, not run on hardware
//
#include <hip/hip_runtime.h>
#include <math.h>

// ---------------------------------------------------------------------------
// CDNA5 (gfx1250) wave32 WMMA bf16 implementation of GatedGraphConvolution.
// All GEMMs via v_wmma_f32_16x16x32_bf16; branchless activations on the raw
// HW transcendentals; 32 edges per wave so each weight (B) fragment feeds two
// WMMAs; per-head deferred shuffle reduction for the attention scores.
// ---------------------------------------------------------------------------

typedef __bf16 bf16;
typedef __attribute__((ext_vector_type(16))) __bf16 v16bf;
typedef __attribute__((ext_vector_type(8)))  __bf16 v8bf;
typedef __attribute__((ext_vector_type(8)))  float   v8f;

#define DD      64
#define LD_FE   200   // padded leading dim for 32x(<=192) bf16 A tiles
#define LD_PW   72    // padded leading dim for 32x64 pw*gate tile

__device__ __forceinline__ v8f wmma_bf(v16bf a, v16bf b, v8f c) {
  return __builtin_amdgcn_wmma_f32_16x16x32_bf16(false, a, false, b,
                                                 (short)0, c, false, false);
}

// A fragment (16x32 bf16, M x K) from a row-major LDS tile.
// lane l: row m=l&15; K = kblock + (l>>4)*16 + {0..15} in v16bf[0..15].
// -> two ds_load_b128 per fragment.
__device__ __forceinline__ v16bf lda(const bf16* tile, int ld, int kblock,
                                     int lane) {
  const int m  = lane & 15;
  const int kb = kblock + ((lane >> 4) << 4);
  const bf16* p = tile + m * ld + kb;
  v8bf lo = *(const v8bf*)p;
  v8bf hi = *(const v8bf*)(p + 8);
  v16bf r;
#pragma unroll
  for (int j = 0; j < 8; ++j) { r[j] = lo[j]; r[j + 8] = hi[j]; }
  return r;
}

// B fragment (32x16 bf16, K x N) from transposed weight Wt[N][K] in global.
// lane l: column n=l&15; K = kblock + (l>>4)*16 + {0..15} -> 32B contiguous.
__device__ __forceinline__ v16bf ldb(const bf16* __restrict__ wt, int K,
                                     int ntile, int kblock, int lane) {
  const int n  = (ntile << 4) + (lane & 15);
  const int k0 = kblock + ((lane >> 4) << 4);
  return *(const v16bf*)(wt + (size_t)n * K + k0);
}

// --- branchless activations on raw HW transcendentals -----------------------
__device__ __forceinline__ float fast_exp(float x) {   // e^x
  return __builtin_amdgcn_exp2f(x * 1.4426950408889634f);
}
__device__ __forceinline__ float fast_ln(float x) {    // ln(x), x >= 1 here
  return __builtin_amdgcn_logf(x) * 0.6931471805599453f;
}
__device__ __forceinline__ float softplusf(float x) {
  // max(x,0) + log1p(exp(-|x|)): no overflow, no branches.
  return fmaxf(x, 0.0f) + fast_ln(1.0f + fast_exp(-fabsf(x)));
}
__device__ __forceinline__ float sigmoidf(float x) {
  return __builtin_amdgcn_rcpf(1.0f + fast_exp(-x));
}
__device__ __forceinline__ float eluf(float x) {
  // x>0: x + (e^0-1)=x ; x<=0: 0 + (e^x-1)
  return fmaxf(x, 0.0f) + (fast_exp(fminf(x, 0.0f)) - 1.0f);
}
__device__ __forceinline__ void st4(bf16* o, float4 v) {
  o[0] = (bf16)v.x; o[1] = (bf16)v.y; o[2] = (bf16)v.z; o[3] = (bf16)v.w;
}

struct alignas(16) WaveBuf {
  bf16  fe[32 * LD_FE];   // staging for final_fe / cat / plane_wave (bf16)
  bf16  pwg[32 * LD_PW];  // plane_wave * gate
  float alpha[128];       // [32 rows][4 heads]
  int   sIdx[32];
  int   tIdx[32];
  float rinv[32];
};

// ---------------------------------------------------------------------------
// Weight prep: f32 W[K][N] -> bf16 Wt[N][K]
// ---------------------------------------------------------------------------
__global__ void transpose_bf16_kernel(const float* __restrict__ W,
                                      bf16* __restrict__ Wt, int K, int N) {
  const int i = blockIdx.x * blockDim.x + threadIdx.x;
  if (i >= K * N) return;
  const int k = i / N, n = i - k * N;
  Wt[(size_t)n * K + k] = (bf16)W[i];
}

// ---------------------------------------------------------------------------
// Pass 1: alpha[e,h] pre-softmax values + atomicMax per (src,h).
// 2 waves/block, 32 edges/wave (two 16-row A tiles share each B fragment).
// ---------------------------------------------------------------------------
__global__ __launch_bounds__(64) void agat_alpha_kernel(
    const float* __restrict__ input, const float* __restrict__ edge_attr,
    const bf16* __restrict__ WaT, const float* __restrict__ att,
    const float* __restrict__ gamma, const float* __restrict__ beta,
    const int* __restrict__ src, const int* __restrict__ tgt,
    float* __restrict__ alphaOut, unsigned int* __restrict__ segMax, int nE) {
  __shared__ WaveBuf sb[2];
  const int wave = threadIdx.x >> 5;
  const int lane = threadIdx.x & 31;
  const int half = lane >> 4;
  WaveBuf& L = sb[wave];
  const int e0 = (blockIdx.x * 2 + wave) * 32;

  {
    int e = e0 + lane; if (e >= nE) e = nE - 1;
    L.sIdx[lane] = src[e];
    L.tIdx[lane] = tgt[e];
  }
#pragma unroll
  for (int j = 0; j < 4; ++j) L.alpha[lane * 4 + j] = 0.0f;
  __syncthreads();

  // cat_i = [input[src] | edge_attr]  (32 x 128 bf16)
  for (int i = lane; i < 32 * 32; i += 32) {
    const int r = i >> 5;
    const int c = (i & 31) << 2;
    int e = e0 + r; if (e >= nE) e = nE - 1;
    const float* p = (c < 64)
        ? (input + (size_t)L.sIdx[r] * DD + c)
        : (edge_attr + (size_t)e * DD + (c - 64));
    st4(&L.fe[r * LD_FE + c], *(const float4*)p);
  }
  __syncthreads();

#pragma unroll 1
  for (int pass = 0; pass < 2; ++pass) {
    const int attOff = pass * 64;
#pragma unroll 1
    for (int h = 0; h < 4; ++h) {
      float p0[8], p1[8];
#pragma unroll
      for (int v = 0; v < 8; ++v) { p0[v] = 0.0f; p1[v] = 0.0f; }
#pragma unroll 1
      for (int dt = 0; dt < 4; ++dt) {
        const int nt = (h << 2) + dt;
        v8f acc0 = {}, acc1 = {};
#pragma unroll
        for (int kb = 0; kb < 128; kb += 32) {
          const v16bf b = ldb(WaT, 128, nt, kb, lane);
          acc0 = wmma_bf(lda(L.fe, LD_FE, kb, lane), b, acc0);
          acc1 = wmma_bf(lda(L.fe + 16 * LD_FE, LD_FE, kb, lane), b, acc1);
        }
        const float attv = att[h * 128 + attOff + (dt << 4) + (lane & 15)];
#pragma unroll
        for (int v = 0; v < 8; ++v) {
          p0[v] += softplusf(acc0[v]) * attv;
          p1[v] += softplusf(acc1[v]) * attv;
        }
      }
      // one 16-lane butterfly reduction per head (columns -> row sums)
#pragma unroll
      for (int v = 0; v < 8; ++v) {
        float x0 = p0[v], x1 = p1[v];
        x0 += __shfl_xor(x0, 1); x1 += __shfl_xor(x1, 1);
        x0 += __shfl_xor(x0, 2); x1 += __shfl_xor(x1, 2);
        x0 += __shfl_xor(x0, 4); x1 += __shfl_xor(x1, 4);
        x0 += __shfl_xor(x0, 8); x1 += __shfl_xor(x1, 8);
        if ((lane & 15) == 0) {
          L.alpha[(v + (half << 3)) * 4 + h]      += x0;  // rows 0..15
          L.alpha[(16 + v + (half << 3)) * 4 + h] += x1;  // rows 16..31
        }
      }
    }
    __syncthreads();
    if (pass == 0) {  // swap node part to targets -> cat_j
      for (int i = lane; i < 32 * 16; i += 32) {
        const int r = i >> 4;
        const int c = (i & 15) << 2;
        st4(&L.fe[r * LD_FE + c],
            *(const float4*)(input + (size_t)L.tIdx[r] * DD + c));
      }
      __syncthreads();
    }
  }

  if ((e0 + lane) < nE) {
    const int e = e0 + lane;
    const float invs = rsqrtf(1.0f + 1e-5f);
#pragma unroll
    for (int h = 0; h < 4; ++h) {
      const float a1 = softplusf(L.alpha[lane * 4 + h]);
      const float a2 = softplusf(a1 * invs * gamma[h] + beta[h]);
      alphaOut[(size_t)e * 4 + h] = a2;
      // a2 = softplus(..) > 0, so uint-ordered atomicMax is correct.
      atomicMax(&segMax[(size_t)L.sIdx[lane] * 4 + h], __float_as_uint(a2));
    }
  }
}

// ---------------------------------------------------------------------------
// Pass 2: a = exp(alpha - segmax[src]); segsum[src] += a
// ---------------------------------------------------------------------------
__global__ void seg_exp_sum_kernel(const int* __restrict__ src,
                                   const unsigned int* __restrict__ segMax,
                                   float* __restrict__ alphaA,
                                   float* __restrict__ segSum, int nE) {
  const int i = blockIdx.x * blockDim.x + threadIdx.x;
  if (i >= nE * 4) return;
  const int e = i >> 2, h = i & 3;
  const int s = src[e];
  const float m = __uint_as_float(segMax[(size_t)s * 4 + h]);
  const float a = fast_exp(alphaA[i] - m);
  alphaA[i] = a;
  atomicAdd(&segSum[(size_t)s * 4 + h], a);
}

// ---------------------------------------------------------------------------
// Pass 3: z2 (plane-wave), z1 (out_j * alpha mean), egm (gate*elu), scatter.
// ---------------------------------------------------------------------------
__global__ __launch_bounds__(64) void fuse_scatter_kernel(
    const float* __restrict__ input, const float* __restrict__ rij,
    const float* __restrict__ plane_wave, const float* __restrict__ edge_attr,
    const bf16* __restrict__ WgT, const bf16* __restrict__ WmT,
    const bf16* __restrict__ W2gT, const bf16* __restrict__ W2T,
    const bf16* __restrict__ WaT, const float* __restrict__ alphaA,
    const float* __restrict__ segSum, const int* __restrict__ src,
    const int* __restrict__ tgt, float* __restrict__ out, int nE) {
  __shared__ WaveBuf sb[2];
  const int wave = threadIdx.x >> 5;
  const int lane = threadIdx.x & 31;
  const int half = lane >> 4;
  WaveBuf& L = sb[wave];
  const int e0 = (blockIdx.x * 2 + wave) * 32;

  {
    int e = e0 + lane; if (e >= nE) e = nE - 1;
    const int s = src[e];
    L.sIdx[lane] = s;
    L.tIdx[lane] = tgt[e];
    float r = rij[e];
    if (r == 0.0f) r = 1e-8f;
    L.rinv[lane] = __builtin_amdgcn_rcpf(r);
#pragma unroll
    for (int h = 0; h < 4; ++h) {
      const float a   = alphaA[(size_t)e * 4 + h];
      const float den = segSum[(size_t)s * 4 + h] + 1e-16f;
      L.alpha[lane * 4 + h] = a * __builtin_amdgcn_rcpf(den);
    }
  }
  __syncthreads();

  // ---- plane-wave branch: z2 = (pw * sigmoid(pw @ W2g)) @ W2 ----
  for (int i = lane; i < 32 * 16; i += 32) {
    const int r = i >> 4;
    const int c = (i & 15) << 2;
    int e = e0 + r; if (e >= nE) e = nE - 1;
    st4(&L.fe[r * LD_FE + c],
        *(const float4*)(plane_wave + (size_t)e * 64 + c));
  }
  __syncthreads();

#pragma unroll 1
  for (int nt = 0; nt < 4; ++nt) {
    v8f acc0 = {}, acc1 = {};
#pragma unroll
    for (int kb = 0; kb < 64; kb += 32) {
      const v16bf b = ldb(W2gT, 64, nt, kb, lane);
      acc0 = wmma_bf(lda(L.fe, LD_FE, kb, lane), b, acc0);
      acc1 = wmma_bf(lda(L.fe + 16 * LD_FE, LD_FE, kb, lane), b, acc1);
    }
    const int n = (nt << 4) + (lane & 15);
#pragma unroll
    for (int v = 0; v < 8; ++v) {
      const int m0 = v + (half << 3);
      const int m1 = 16 + m0;
      L.pwg[m0 * LD_PW + n] =
          (bf16)(sigmoidf(acc0[v]) * (float)L.fe[m0 * LD_FE + n]);
      L.pwg[m1 * LD_PW + n] =
          (bf16)(sigmoidf(acc1[v]) * (float)L.fe[m1 * LD_FE + n]);
    }
  }
  __syncthreads();

  v8f zacc0[4], zacc1[4];
#pragma unroll 1
  for (int dt = 0; dt < 4; ++dt) {
    v8f acc0 = {}, acc1 = {};
#pragma unroll
    for (int kb = 0; kb < 64; kb += 32) {
      const v16bf b = ldb(W2T, 64, dt, kb, lane);
      acc0 = wmma_bf(lda(L.pwg, LD_PW, kb, lane), b, acc0);
      acc1 = wmma_bf(lda(L.pwg + 16 * LD_PW, LD_PW, kb, lane), b, acc1);
    }
    zacc0[dt] = acc0;                     // zacc starts as z2
    zacc1[dt] = acc1;
  }
  __syncthreads();

  // ---- AGAT z1: out_j = softplus([input[tgt]|edge_attr] @ Wa) ----
  for (int i = lane; i < 32 * 32; i += 32) {
    const int r = i >> 5;
    const int c = (i & 31) << 2;
    int e = e0 + r; if (e >= nE) e = nE - 1;
    const float* p = (c < 64)
        ? (input + (size_t)L.tIdx[r] * DD + c)
        : (edge_attr + (size_t)e * DD + (c - 64));
    st4(&L.fe[r * LD_FE + c], *(const float4*)p);
  }
  __syncthreads();

#pragma unroll 1
  for (int nt = 0; nt < 16; ++nt) {
    v8f acc0 = {}, acc1 = {};
#pragma unroll
    for (int kb = 0; kb < 128; kb += 32) {
      const v16bf b = ldb(WaT, 128, nt, kb, lane);
      acc0 = wmma_bf(lda(L.fe, LD_FE, kb, lane), b, acc0);
      acc1 = wmma_bf(lda(L.fe + 16 * LD_FE, LD_FE, kb, lane), b, acc1);
    }
    const int h  = nt >> 2;   // 16 cols per tile, 64 per head
    const int dt = nt & 3;    // output d-tile
#pragma unroll
    for (int v = 0; v < 8; ++v) {
      const int m0 = v + (half << 3);
      zacc0[dt][v] += softplusf(acc0[v]) * L.alpha[m0 * 4 + h] * 0.25f;
      zacc1[dt][v] += softplusf(acc1[v]) * L.alpha[(16 + m0) * 4 + h] * 0.25f;
    }
  }
  __syncthreads();

  // ---- gated edge MLP: final_fe = [ni | nj | (ni-nj)/r] (32 x 192) ----
  for (int i = lane; i < 32 * 16; i += 32) {
    const int r = i >> 4;
    const int c = (i & 15) << 2;
    const float4 a = *(const float4*)(input + (size_t)L.sIdx[r] * DD + c);
    const float4 b = *(const float4*)(input + (size_t)L.tIdx[r] * DD + c);
    const float ri = L.rinv[r];
    bf16* o = &L.fe[r * LD_FE];
    st4(o + c, a);
    st4(o + 64 + c, b);
    float4 d;
    d.x = (a.x - b.x) * ri; d.y = (a.y - b.y) * ri;
    d.z = (a.z - b.z) * ri; d.w = (a.w - b.w) * ri;
    st4(o + 128 + c, d);
  }
  __syncthreads();

#pragma unroll 1
  for (int dt = 0; dt < 4; ++dt) {
    v8f g0 = {}, m0a = {}, g1 = {}, m1a = {};
#pragma unroll
    for (int kb = 0; kb < 192; kb += 32) {
      const v16bf a0 = lda(L.fe, LD_FE, kb, lane);
      const v16bf a1 = lda(L.fe + 16 * LD_FE, LD_FE, kb, lane);
      const v16bf bg = ldb(WgT, 192, dt, kb, lane);
      const v16bf bm = ldb(WmT, 192, dt, kb, lane);
      g0  = wmma_bf(a0, bg, g0);
      g1  = wmma_bf(a1, bg, g1);
      m0a = wmma_bf(a0, bm, m0a);
      m1a = wmma_bf(a1, bm, m1a);
    }
    const int col = (dt << 4) + (lane & 15);
#pragma unroll
    for (int v = 0; v < 8; ++v) {
      const int r0 = v + (half << 3);
      const int r1 = 16 + r0;
      if (e0 + r0 < nE) {
        const float z = sigmoidf(g0[v]) * eluf(m0a[v]) * zacc0[dt][v];
        atomicAdd(&out[(size_t)L.sIdx[r0] * DD + col], z);
      }
      if (e0 + r1 < nE) {
        const float z = sigmoidf(g1[v]) * eluf(m1a[v]) * zacc1[dt][v];
        atomicAdd(&out[(size_t)L.sIdx[r1] * DD + col], z);
      }
    }
  }
}

// ---------------------------------------------------------------------------
extern "C" void kernel_launch(void* const* d_in, const int* in_sizes, int n_in,
                              void* d_out, int out_size, void* d_ws,
                              size_t ws_size, hipStream_t stream) {
  const float* input = (const float*)d_in[0];
  const float* rij   = (const float*)d_in[1];
  const float* pw    = (const float*)d_in[2];
  const float* eat   = (const float*)d_in[3];
  const float* Wg    = (const float*)d_in[4];
  const float* Wm    = (const float*)d_in[5];
  const float* W2    = (const float*)d_in[6];
  const float* W2g   = (const float*)d_in[7];
  const float* Wa    = (const float*)d_in[8];
  const float* att   = (const float*)d_in[9];
  const float* gam   = (const float*)d_in[10];
  const float* bet   = (const float*)d_in[11];
  const int*   src   = (const int*)d_in[12];
  const int*   tgt   = (const int*)d_in[13];
  // d_in[14] = edge_index (duplicate of src/tgt), d_in[15] = nodes (unused)

  const int nE = in_sizes[1];            // 640000
  const int nN = in_sizes[0] / DD;       // 40000

  // Workspace layout (bf16 transposed weights, then per-edge/per-node scratch)
  char* ws = (char*)d_ws;
  bf16* WgT  = (bf16*)(ws + 0);          // 64  x 192 -> 24576 B
  bf16* WmT  = (bf16*)(ws + 24576);      // 64  x 192 -> 24576 B
  bf16* W2gT = (bf16*)(ws + 49152);      // 64  x 64  ->  8192 B
  bf16* W2T  = (bf16*)(ws + 57344);      // 64  x 64  ->  8192 B
  bf16* WaT  = (bf16*)(ws + 65536);      // 256 x 128 -> 65536 B
  float* alphaA = (float*)(ws + 131072);                       // E*4 f32
  size_t off = 131072 + (size_t)nE * 4 * sizeof(float);
  unsigned int* segMax = (unsigned int*)(ws + off);            // N*4
  float* segSum = (float*)(ws + off + (size_t)nN * 4 * sizeof(float));

  // out = input (scatter-add target); zero segment max/sum (contiguous)
  hipMemcpyAsync(d_out, (const void*)input, (size_t)in_sizes[0] * sizeof(float),
                 hipMemcpyDeviceToDevice, stream);
  hipMemsetAsync(segMax, 0, 2 * (size_t)nN * 4 * sizeof(float), stream);

  // Weight prep: f32 [K,N] -> bf16 [N,K]
  transpose_bf16_kernel<<<(192 * 64 + 255) / 256, 256, 0, stream>>>(Wg, WgT, 192, 64);
  transpose_bf16_kernel<<<(192 * 64 + 255) / 256, 256, 0, stream>>>(Wm, WmT, 192, 64);
  transpose_bf16_kernel<<<(64 * 64 + 255) / 256, 256, 0, stream>>>(W2g, W2gT, 64, 64);
  transpose_bf16_kernel<<<(64 * 64 + 255) / 256, 256, 0, stream>>>(W2, W2T, 64, 64);
  transpose_bf16_kernel<<<(128 * 256 + 255) / 256, 256, 0, stream>>>(Wa, WaT, 128, 256);

  const int blocks = (nE + 63) / 64;     // 2 waves/block, 32 edges/wave
  agat_alpha_kernel<<<blocks, 64, 0, stream>>>(
      input, eat, WaT, att, gam, bet, src, tgt, alphaA, segMax, nE);
  seg_exp_sum_kernel<<<(nE * 4 + 255) / 256, 256, 0, stream>>>(
      src, segMax, alphaA, segSum, nE);
  fuse_scatter_kernel<<<blocks, 64, 0, stream>>>(
      input, rij, pw, eat, WgT, WmT, W2gT, W2T, WaT, alphaA, segSum, src, tgt,
      (float*)d_out, nE);
}
